// CrossAttentionS2T_88441966559400
// MI455X (gfx1250) — matrix operand order repaired and checked by reading, not compiled, not run on hardware
//
#include <hip/hip_runtime.h>

// ---------------------------------------------------------------------------
// CrossAttention S->T for MI455X (gfx1250, wave32, WMMA bf16 16x16x32).
//
// Pipeline: [cvt weights f32->bf16] -> [Q gemm (+pos)] -> [KV gemm (+pos,
// V stored transposed)] -> [fused flash attention, 1 wave / (bt,h,qtile)] ->
// [output projection -> f32 d_out].
// Matrix math: v_wmma_f32_16x16x32_bf16, f32 accumulate.
// Data movement: B weight tiles via TDM (tensor_load_to_lds, TENSORcnt),
// bf16 activation tiles via GLOBAL_LOAD_ASYNC_TO_LDS (ASYNCcnt).
// ---------------------------------------------------------------------------

typedef __bf16 bf16;
typedef __attribute__((ext_vector_type(16))) __bf16 bf16x16;
typedef __attribute__((ext_vector_type(8)))  __bf16 bf16x8;
typedef __attribute__((ext_vector_type(4)))  __bf16 bf16x4;
typedef __attribute__((ext_vector_type(8)))  float  f32x8;
typedef int          v4i   __attribute__((vector_size(4 * sizeof(int))));
typedef unsigned int u32x4 __attribute__((vector_size(16)));
typedef int          i32x4 __attribute__((vector_size(16)));
typedef int          i32x8 __attribute__((vector_size(32)));

#define BT_    128
#define NTOK_  196
#define DIM_   768
#define NH_    12
#define HD_    64
#define NPAD_  224            // 7 * 32 padded key/value token count
#define MROWS_ (BT_ * NTOK_)  // 25088
#define SCALE_ 0.125f         // 64^-0.5

#define WMMA_BF16(A, B, C) \
  __builtin_amdgcn_wmma_f32_16x16x32_bf16(false, (A), false, (B), (short)0, (C), false, false)

// ---- Async global -> LDS (ASYNCcnt) --------------------------------------
#if defined(__has_builtin)
#if __has_builtin(__builtin_amdgcn_global_load_async_to_lds_b128)
#define ASYNC_LDS 1
#endif
#if __has_builtin(__builtin_amdgcn_tensor_load_to_lds) && \
    __has_builtin(__builtin_amdgcn_s_wait_tensorcnt)
#define HAVE_TDM 1
#endif
#endif

#if defined(ASYNC_LDS)
// param 0: v4i addrspace(1)* (global src), param 1: v4i addrspace(3)* (LDS dst)
#define GPTR(p) ((__attribute__((address_space(1))) v4i*)(unsigned long long)(const void*)(p))
#define LPTR(p) ((__attribute__((address_space(3))) v4i*)(void*)(p))
#if __has_builtin(__builtin_amdgcn_s_wait_asynccnt)
#define WAIT_ASYNC() __builtin_amdgcn_s_wait_asynccnt(0)
#else
#define WAIT_ASYNC() asm volatile("s_wait_asynccnt 0" ::: "memory")
#endif
#else
#define WAIT_ASYNC()
#endif

// LDS byte address (wave-relative) of a __shared__ object
static __device__ __forceinline__ unsigned lds_addr_of(void* p) {
  return (unsigned)(size_t)(__attribute__((address_space(3))) void*)p;
}

#if defined(HAVE_TDM)
// Issue a 2D TDM load: tile_rows x 64B tile from a row-major bf16 tensor with
// row stride 1536B into LDS rows padded to 80B (pad 4 dwords per 16 dwords).
// D# built per CDNA5 ISA 8.3-8.5; data_size = 4B units.
static __device__ __forceinline__ void tdm_load_tile(const bf16* gsrc,
                                                     unsigned lds_byte_addr,
                                                     int tile_rows,
                                                     int tensor_rows) {
  unsigned long long ga = (unsigned long long)gsrc;
  u32x4 g0 = {
      1u,                                              // count=1, user D#
      lds_byte_addr,                                   // lds_addr
      (unsigned)(ga & 0xFFFFFFFFull),                  // global_addr[31:0]
      (unsigned)((ga >> 32) & 0x1FFFFFFull) | (2u << 30)  // addr[56:32]|type=2
  };
  i32x8 g1 = {
      (int)((2u << 16) | (1u << 20) | (3u << 22) | (3u << 25)),
      //    data_size=4B  pad_en      pad_int=16dw  pad_amt=4dw
      (int)(384u << 16),                // tensor_dim0 = 384 dwords (lo16)
      (int)(((unsigned)tensor_rows & 0xFFFFu) << 16),  // tensor_dim1 lo16
      (int)(16u << 16),                 // tile_dim0 = 16 dwords
      tile_rows,                        // tile_dim1 ; tile_dim2=0
      384,                              // tensor_dim0_stride lo32 (dwords)
      0, 0                              // stride hi / dim1_stride (unused 2D)
  };
  i32x4 z4 = {0, 0, 0, 0};
#if __clang_major__ >= 23
  i32x8 z8 = {0, 0, 0, 0, 0, 0, 0, 0};
  __builtin_amdgcn_tensor_load_to_lds(g0, g1, z4, z4, z8, 0);
#else
  __builtin_amdgcn_tensor_load_to_lds(g0, g1, z4, z4, 0);
#endif
}
#endif

// A-operand fragment (16x32 bf16, MxK): lane holds row M=lane%16 with the
// 16 contiguous K values K = 16h .. 16h+15, h=lane/16. p points at element K=16h.
static __device__ __forceinline__ bf16x16 ld_frag_a(const bf16* p) {
  bf16x8 lo = *(const bf16x8*)(p);
  bf16x8 hi = *(const bf16x8*)(p + 8);
  bf16x16 r;
#pragma unroll
  for (int e = 0; e < 8; ++e) { r[e] = lo[e]; r[e + 8] = hi[e]; }
  return r;
}

// B-operand fragment (32x16 bf16, KxN): lane holds column N=lane%16 with the
// 16 contiguous K values K = 16*(lane/16) .. +15. p points at first element.
static __device__ __forceinline__ bf16x16 ld_frag_b(const bf16* p) {
  bf16x8 lo = *(const bf16x8*)(p);
  bf16x8 hi = *(const bf16x8*)(p + 8);
  bf16x16 r;
#pragma unroll
  for (int e = 0; e < 8; ++e) { r[e] = lo[e]; r[e + 8] = hi[e]; }
  return r;
}

// ---------------------------------------------------------------------------
// Weight conversion f32 -> bf16 (vector-of-4 per thread)
// ---------------------------------------------------------------------------
__global__ void cvt_f32_bf16(const float* __restrict__ src,
                             bf16* __restrict__ dst, int n4) {
  int i = blockIdx.x * 256 + threadIdx.x;
  if (i < n4) {
    float4 v = ((const float4*)src)[i];
    bf16x4 o;
    o[0] = (bf16)v.x; o[1] = (bf16)v.y; o[2] = (bf16)v.z; o[3] = (bf16)v.w;
    ((bf16x4*)dst)[i] = o;
  }
}

// ---------------------------------------------------------------------------
// GEMM: out[M x Nout] = act(X)[M x 768] @ W[Nout x 768]^T + bias
// IN_MODE 0: X = t_x fp32 + vmae pos     (rows = bt*196 + token)
// IN_MODE 1: X = s_x fp32 (drop CLS) + clip pos
// IN_MODE 2: X = bf16 activations, no pos (async-LDS staged)
// OUT_MODE 0: bf16 plain | 1: K plain / V transposed | 2: f32 plain
// Block: 256 threads (8 waves, 4x2), tile 128(M) x 64(N), K-step 32.
// B weight tile staged by TDM (one descriptor per k-step, wave 0).
// ---------------------------------------------------------------------------
template <int IN_MODE, int OUT_MODE>
__global__ __launch_bounds__(256)
void gemm_bf16(const float* __restrict__ Xf, const bf16* __restrict__ Xb,
               const float* __restrict__ pos, const bf16* __restrict__ Wb,
               const float* __restrict__ bias,
               bf16* __restrict__ outb, bf16* __restrict__ kout,
               bf16* __restrict__ vt, float* __restrict__ outf) {
  __shared__ __align__(16) bf16 As[128][40];  // 80B row pitch (bank spread)
  __shared__ __align__(16) bf16 Bs[64][40];

  const int tid  = threadIdx.x;
  const int lane = tid & 31;
  const int wave = tid >> 5;
  const int wm = wave >> 1, wn = wave & 1;   // 4 x 2 waves
  const int cl = lane & 15, half = lane >> 4;
  const int m0 = blockIdx.y * 128;
  const int n0 = blockIdx.x * 64;
  const int nrows = (int)gridDim.x * 64;     // total weight rows

  f32x8 acc[2][2];
#pragma unroll
  for (int i = 0; i < 2; ++i)
#pragma unroll
    for (int j = 0; j < 2; ++j)
#pragma unroll
      for (int e = 0; e < 8; ++e) acc[i][j][e] = 0.f;

  const int arow = tid >> 3;        // 0..31
  const int ac4  = (tid & 7) * 4;   // 0,4,..,28

  for (int kk = 0; kk < DIM_; kk += 32) {
    // ---- stage B tile (64 weight rows x 32 k) ----
#if defined(HAVE_TDM)
    if (wave == 0) {
      tdm_load_tile(Wb + (size_t)n0 * DIM_ + kk, lds_addr_of(&Bs[0][0]),
                    64, nrows);
      __builtin_amdgcn_s_wait_tensorcnt(0);
    }
#else
    {
      int r = tid >> 2, ch = tid & 3;
      const bf16* src = Wb + (size_t)(n0 + r) * DIM_ + kk + ch * 8;
#if defined(ASYNC_LDS)
      __builtin_amdgcn_global_load_async_to_lds_b128(GPTR(src),
                                                     LPTR(&Bs[r][ch * 8]), 0, 0);
#else
      *(bf16x8*)&Bs[r][ch * 8] = *(const bf16x8*)src;
#endif
    }
#endif
    // ---- stage A tile (128 x 32) into LDS as bf16 ----
    if (IN_MODE == 2) {
#pragma unroll
      for (int j = 0; j < 2; ++j) {
        int id = tid + 256 * j;
        int r = id >> 2, ch = id & 3;
        const bf16* src = Xb + (size_t)(m0 + r) * DIM_ + kk + ch * 8;
#if defined(ASYNC_LDS)
        __builtin_amdgcn_global_load_async_to_lds_b128(GPTR(src),
                                                       LPTR(&As[r][ch * 8]), 0, 0);
#else
        *(bf16x8*)&As[r][ch * 8] = *(const bf16x8*)src;
#endif
      }
    } else {
#pragma unroll
      for (int i = 0; i < 4; ++i) {
        int r = arow + 32 * i;
        int gm = m0 + r;
        int token = gm % NTOK_;
        const float* srcp;
        if (IN_MODE == 0) {
          srcp = Xf + (size_t)gm * DIM_ + kk + ac4;
        } else {
          int b = gm / NTOK_;
          srcp = Xf + ((size_t)(b * (NTOK_ + 1) + 1 + token)) * DIM_ + kk + ac4;
        }
        float4 x  = *(const float4*)srcp;
        float4 pz = *(const float4*)(pos + (size_t)token * DIM_ + kk + ac4);
        As[r][ac4 + 0] = (bf16)(x.x + pz.x);
        As[r][ac4 + 1] = (bf16)(x.y + pz.y);
        As[r][ac4 + 2] = (bf16)(x.z + pz.z);
        As[r][ac4 + 3] = (bf16)(x.w + pz.w);
      }
    }
    WAIT_ASYNC();
    __syncthreads();

    bf16x16 afr[2], bfr[2];
#pragma unroll
    for (int mt = 0; mt < 2; ++mt)
      afr[mt] = ld_frag_a(&As[wm * 32 + mt * 16 + cl][16 * half]);
#pragma unroll
    for (int nt = 0; nt < 2; ++nt)
      bfr[nt] = ld_frag_b(&Bs[wn * 32 + nt * 16 + cl][16 * half]);
#pragma unroll
    for (int mt = 0; mt < 2; ++mt)
#pragma unroll
      for (int nt = 0; nt < 2; ++nt)
        acc[mt][nt] = WMMA_BF16(afr[mt], bfr[nt], acc[mt][nt]);
    __syncthreads();
  }

  // ---- epilogue: bias + store ----
#pragma unroll
  for (int mt = 0; mt < 2; ++mt) {
#pragma unroll
    for (int nt = 0; nt < 2; ++nt) {
      int gn = n0 + wn * 32 + nt * 16 + cl;
      float bv = bias[gn];
#pragma unroll
      for (int r = 0; r < 8; ++r) {
        int gm = m0 + wm * 32 + mt * 16 + r + 8 * half;
        float val = acc[mt][nt][r] + bv;
        if (OUT_MODE == 0) {
          outb[(size_t)gm * DIM_ + gn] = (bf16)val;
        } else if (OUT_MODE == 2) {
          outf[(size_t)gm * DIM_ + gn] = val;
        } else {
          if (gn < DIM_) {
            kout[(size_t)gm * DIM_ + gn] = (bf16)val;
          } else {
            int c = gn - DIM_;
            int h = c >> 6, d = c & 63;
            int b = gm / NTOK_, token = gm % NTOK_;
            vt[(((size_t)b * NH_ + h) * HD_ + d) * NPAD_ + token] = (bf16)val;
          }
        }
      }
    }
  }
}

// ---------------------------------------------------------------------------
// Fused flash attention. One wave per (q-tile of 32 rows, head, bt).
// S = (Q Kt) * SCALE, online softmax, O += P V. V is pre-transposed
// [bt][h][d][token(224)], padding zeroed.
// ---------------------------------------------------------------------------
__global__ __launch_bounds__(32)
void attn_fused(const bf16* __restrict__ qb, const bf16* __restrict__ kb,
                const bf16* __restrict__ vt, bf16* __restrict__ ob) {
  __shared__ __align__(16) bf16 Pm[32][40];
  const int lane = threadIdx.x;
  const int cl = lane & 15, half = lane >> 4;
  const int qt = blockIdx.x, h = blockIdx.y, bt = blockIdx.z;
  const int q0 = qt * 32;

  // Resident Q fragments: 2 m-tiles x 2 k-chunks of 32
  bf16x16 qf[2][2];
#pragma unroll
  for (int mt = 0; mt < 2; ++mt) {
    int token = q0 + mt * 16 + cl;
    if (token > NTOK_ - 1) token = NTOK_ - 1;   // clamp (rows masked at store)
    const bf16* base = qb + ((size_t)(bt * NTOK_ + token)) * DIM_ + h * HD_;
#pragma unroll
    for (int ks = 0; ks < 2; ++ks)
      qf[mt][ks] = ld_frag_a(base + ks * 32 + 16 * half);
  }

  float mrow[2][8], lrow[2][8];
  f32x8 oacc[2][4];
#pragma unroll
  for (int mt = 0; mt < 2; ++mt) {
#pragma unroll
    for (int r = 0; r < 8; ++r) { mrow[mt][r] = -1e30f; lrow[mt][r] = 0.f; }
#pragma unroll
    for (int nd = 0; nd < 4; ++nd)
#pragma unroll
      for (int e = 0; e < 8; ++e) oacc[mt][nd][e] = 0.f;
  }

  for (int kc = 0; kc < 7; ++kc) {
    // prefetch next chunk of K rows / V rows into cache (global_prefetch_b8)
    if (kc + 1 < 7) {
      int tokp = (kc + 1) * 32 + lane;
      if (tokp > NTOK_ - 1) tokp = NTOK_ - 1;
      __builtin_prefetch(kb + ((size_t)(bt * NTOK_ + tokp)) * DIM_ + h * HD_, 0, 1);
      const bf16* vbase = vt + ((size_t)(bt * NH_ + h)) * HD_ * NPAD_;
      __builtin_prefetch(vbase + (size_t)lane * NPAD_ + (kc + 1) * 32, 0, 1);
      __builtin_prefetch(vbase + (size_t)(lane + 32) * NPAD_ + (kc + 1) * 32, 0, 1);
    }

    // K^T fragments: column n = key token, contiguous head-dim per lane
    bf16x16 kf[2][2];
#pragma unroll
    for (int nt = 0; nt < 2; ++nt) {
      int tok = kc * 32 + nt * 16 + cl;
      int tc = tok > NTOK_ - 1 ? NTOK_ - 1 : tok;
      const bf16* base = kb + ((size_t)(bt * NTOK_ + tc)) * DIM_ + h * HD_;
#pragma unroll
      for (int ks = 0; ks < 2; ++ks)
        kf[nt][ks] = ld_frag_b(base + ks * 32 + 16 * half);
    }

    // S tiles (32 x 32), scale + column mask
    f32x8 s[2][2];
#pragma unroll
    for (int mt = 0; mt < 2; ++mt)
#pragma unroll
      for (int nt = 0; nt < 2; ++nt) {
        f32x8 z = {0.f, 0.f, 0.f, 0.f, 0.f, 0.f, 0.f, 0.f};
        z = WMMA_BF16(qf[mt][0], kf[nt][0], z);
        z = WMMA_BF16(qf[mt][1], kf[nt][1], z);
        int col = kc * 32 + nt * 16 + cl;
        bool valid = col < NTOK_;
#pragma unroll
        for (int r = 0; r < 8; ++r)
          s[mt][nt][r] = valid ? z[r] * SCALE_ : -1e30f;
      }

    // Online softmax update
#pragma unroll
    for (int mt = 0; mt < 2; ++mt) {
      float nm[8], al[8];
#pragma unroll
      for (int r = 0; r < 8; ++r) {
        float v = fmaxf(s[mt][0][r], s[mt][1][r]);
#pragma unroll
        for (int o = 8; o >= 1; o >>= 1) v = fmaxf(v, __shfl_xor(v, o, 32));
        nm[r] = fmaxf(mrow[mt][r], v);
        al[r] = __expf(mrow[mt][r] - nm[r]);
      }
#pragma unroll
      for (int nt = 0; nt < 2; ++nt)
#pragma unroll
        for (int r = 0; r < 8; ++r) {
          float p = __expf(s[mt][nt][r] - nm[r]);
          s[mt][nt][r] = p;
          Pm[mt * 16 + r + 8 * half][nt * 16 + cl] = (bf16)p;
        }
#pragma unroll
      for (int r = 0; r < 8; ++r) {
        float t = s[mt][0][r] + s[mt][1][r];
#pragma unroll
        for (int o = 8; o >= 1; o >>= 1) t += __shfl_xor(t, o, 32);
        lrow[mt][r] = lrow[mt][r] * al[r] + t;
        mrow[mt][r] = nm[r];
#pragma unroll
        for (int nd = 0; nd < 4; ++nd) oacc[mt][nd][r] *= al[r];
      }
    }
    asm volatile("s_wait_dscnt 0" ::: "memory");

    // O += P x V  (K = 32 chunk tokens)
    bf16x16 pf[2], vf[4];
#pragma unroll
    for (int mt = 0; mt < 2; ++mt)
      pf[mt] = ld_frag_a(&Pm[mt * 16 + cl][16 * half]);
#pragma unroll
    for (int nd = 0; nd < 4; ++nd) {
      const bf16* vb = vt +
          (((size_t)(bt * NH_ + h)) * HD_ + nd * 16 + cl) * NPAD_ +
          kc * 32 + 16 * half;
      vf[nd] = ld_frag_b(vb);
    }
#pragma unroll
    for (int mt = 0; mt < 2; ++mt)
#pragma unroll
      for (int nd = 0; nd < 4; ++nd)
        oacc[mt][nd] = WMMA_BF16(pf[mt], vf[nd], oacc[mt][nd]);
  }

  // normalize + store
#pragma unroll
  for (int mt = 0; mt < 2; ++mt)
#pragma unroll
    for (int r = 0; r < 8; ++r) {
      int token = q0 + mt * 16 + r + 8 * half;
      if (token < NTOK_) {
        float inv = 1.0f / lrow[mt][r];
#pragma unroll
        for (int nd = 0; nd < 4; ++nd)
          ob[((size_t)(bt * NTOK_ + token)) * DIM_ + h * HD_ + nd * 16 + cl] =
              (bf16)(oacc[mt][nd][r] * inv);
      }
    }
}

// ---------------------------------------------------------------------------
// Launcher
// ---------------------------------------------------------------------------
extern "C" void kernel_launch(void* const* d_in, const int* in_sizes, int n_in,
                              void* d_out, int out_size, void* d_ws,
                              size_t ws_size, hipStream_t stream) {
  const float* s_x      = (const float*)d_in[0];
  const float* t_x      = (const float*)d_in[1];
  const float* clip_pos = (const float*)d_in[2];
  const float* vmae_pos = (const float*)d_in[3];
  const float* q_w      = (const float*)d_in[4];
  const float* q_b      = (const float*)d_in[5];
  const float* kv_w     = (const float*)d_in[6];
  const float* kv_b     = (const float*)d_in[7];
  const float* proj_w   = (const float*)d_in[8];
  const float* proj_b   = (const float*)d_in[9];
  float* out = (float*)d_out;

  char* ws = (char*)d_ws;
  size_t off = 0;
  auto wsalloc = [&](size_t bytes) {
    void* p = ws + off;
    off += (bytes + 255) & ~(size_t)255;
    return p;
  };
  bf16* qw_b  = (bf16*)wsalloc((size_t)DIM_ * DIM_ * 2);
  bf16* kvw_b = (bf16*)wsalloc((size_t)2 * DIM_ * DIM_ * 2);
  bf16* pw_b  = (bf16*)wsalloc((size_t)DIM_ * DIM_ * 2);
  bf16* qbuf  = (bf16*)wsalloc((size_t)MROWS_ * DIM_ * 2);
  bf16* kbuf  = (bf16*)wsalloc((size_t)MROWS_ * DIM_ * 2);
  size_t vt_bytes = (size_t)BT_ * NH_ * HD_ * NPAD_ * 2;
  bf16* vtbuf = (bf16*)wsalloc(vt_bytes);
  bf16* obuf  = (bf16*)wsalloc((size_t)MROWS_ * DIM_ * 2);

  // zero V^T (covers token padding so P x V never sees junk)
  (void)hipMemsetAsync(vtbuf, 0, vt_bytes, stream);

  // weights -> bf16
  {
    int n4q = DIM_ * DIM_ / 4;
    int n4kv = 2 * DIM_ * DIM_ / 4;
    cvt_f32_bf16<<<(n4q + 255) / 256, 256, 0, stream>>>(q_w, qw_b, n4q);
    cvt_f32_bf16<<<(n4kv + 255) / 256, 256, 0, stream>>>(kv_w, kvw_b, n4kv);
    cvt_f32_bf16<<<(n4q + 255) / 256, 256, 0, stream>>>(proj_w, pw_b, n4q);
  }

  const int mtiles = MROWS_ / 128;  // 196
  // Q = (t_x + vmae_pos) @ q_w^T + q_b
  gemm_bf16<0, 0><<<dim3(DIM_ / 64, mtiles), 256, 0, stream>>>(
      t_x, nullptr, vmae_pos, qw_b, q_b, qbuf, nullptr, nullptr, nullptr);
  // [K|V] = (s_x[:,1:] + clip_pos) @ kv_w^T + kv_b ; V stored transposed
  gemm_bf16<1, 1><<<dim3(2 * DIM_ / 64, mtiles), 256, 0, stream>>>(
      s_x, nullptr, clip_pos, kvw_b, kv_b, nullptr, kbuf, vtbuf, nullptr);
  // attention
  attn_fused<<<dim3(7, NH_, BT_), 32, 0, stream>>>(qbuf, kbuf, vtbuf, obuf);
  // out = O @ proj_w^T + proj_b (fp32 output)
  gemm_bf16<2, 2><<<dim3(DIM_ / 64, mtiles), 256, 0, stream>>>(
      nullptr, obuf, nullptr, pw_b, proj_b, nullptr, nullptr, nullptr, out);
}
